// GATv2Layer_5007931867340
// MI455X (gfx1250) — matrix operand order repaired and checked by reading, not compiled, not run on hardware
//
#include <hip/hip_runtime.h>
#include <hip/hip_bf16.h>

// ---------------------------------------------------------------------------
// GATv2-style layer for MI455X (gfx1250, wave32).
//   H = x @ W^T          : WMMA bf16 3-pass split-precision GEMM (f32 accum)
//   scores, softmax, agg : edge-parallel kernels, f32 global atomics (L2-bound)
// ---------------------------------------------------------------------------

typedef __attribute__((ext_vector_type(16))) __bf16 v16bf;
typedef __attribute__((ext_vector_type(8)))  float  v8f;

#define NEG_SLOPE 0.2f
#define EPS_F 1e-8f

// ---------------------------------------------------------------------------
// K0: init — zero output accumulator + alpha_sum, set gmax = -inf
// ---------------------------------------------------------------------------
__global__ void k_init(float* __restrict__ out, float* __restrict__ asum,
                       float* __restrict__ gmax, int nOut, int nSum) {
  int i = blockIdx.x * blockDim.x + threadIdx.x;
  int stride = gridDim.x * blockDim.x;
  for (int j = i; j < nOut; j += stride) out[j] = 0.f;
  for (int j = i; j < nSum; j += stride) asum[j] = 0.f;
  if (i == 0) *gmax = -__builtin_inff();
}

// ---------------------------------------------------------------------------
// K1: H = x @ W^T via v_wmma_f32_16x16x32_bf16, split-bf16 (hi/lo) 3-pass.
// Block = 256 threads (8 waves). blockIdx.y = column tile (16 cols of W),
// each wave computes one 16x16 tile of H. W stripe staged in LDS as bf16.
// ---------------------------------------------------------------------------
__global__ void __launch_bounds__(256)
k_gemm_wmma(const float* __restrict__ x, const float* __restrict__ W,
            float* __restrict__ H, int N) {
  __shared__ __align__(32) __bf16 sBhi[16 * 256];
  __shared__ __align__(32) __bf16 sBlo[16 * 256];

  const int colTile = blockIdx.y;  // 0..15

  // Cooperative convert of W[colTile*16 .. +16)[0..256) into bf16 hi/lo.
  for (int idx = threadIdx.x; idx < 16 * 256; idx += 256) {
    const int c = idx >> 8;     // local column 0..15
    const int k = idx & 255;
    const float w = W[(size_t)(colTile * 16 + c) * 256 + k];
    const __bf16 hi = (__bf16)w;
    sBhi[idx] = hi;
    sBlo[idx] = (__bf16)(w - (float)hi);
  }
  __syncthreads();

  const int wave = threadIdx.x >> 5;
  const int lane = threadIdx.x & 31;
  const int nRowTiles = N >> 4;                 // N divisible by 16 (50000)
  const int rowTile = blockIdx.x * 8 + wave;
  if (rowTile >= nRowTiles) return;             // wave-uniform exit (EXEC all-1)

  const int hiHalf = lane >> 4;                 // 0: lanes 0-15, 1: lanes 16-31
  const int row = rowTile * 16 + (lane & 15);   // A: M = lane&15
  const float* __restrict__ xrow = x + (size_t)row * 256;

  v8f acc = {};
  for (int kb = 0; kb < 256; kb += 32) {
    // ---- A fragment: K = kb + hiHalf*16 + (0..15), matching B
    const int ka = kb + hiHalf * 16;
    const float4 f0 = *(const float4*)(xrow + ka);
    const float4 f1 = *(const float4*)(xrow + ka + 4);
    const float4 f2 = *(const float4*)(xrow + ka + 8);
    const float4 f3 = *(const float4*)(xrow + ka + 12);
    const float fa[16] = {f0.x, f0.y, f0.z, f0.w, f1.x, f1.y, f1.z, f1.w,
                          f2.x, f2.y, f2.z, f2.w, f3.x, f3.y, f3.z, f3.w};
    v16bf a_hi, a_lo;
#pragma unroll
    for (int j = 0; j < 16; ++j) {
      const __bf16 h = (__bf16)fa[j];
      a_hi[j] = h;
      a_lo[j] = (__bf16)(fa[j] - (float)h);
    }

    // ---- B fragment from LDS: N = lane&15, K = kb + hiHalf*16 + (0..15)
    const int bOff = (lane & 15) * 256 + kb + hiHalf * 16;
    const v16bf b_hi = *(const v16bf*)(sBhi + bOff);
    const v16bf b_lo = *(const v16bf*)(sBlo + bOff);

    // Split-precision: x*w ~= hi*hi + hi*lo + lo*hi (f32 accumulate)
    acc = __builtin_amdgcn_wmma_f32_16x16x32_bf16(false, a_hi, false, b_hi,
                                                  (short)0, acc, false, false);
    acc = __builtin_amdgcn_wmma_f32_16x16x32_bf16(false, a_hi, false, b_lo,
                                                  (short)0, acc, false, false);
    acc = __builtin_amdgcn_wmma_f32_16x16x32_bf16(false, a_lo, false, b_hi,
                                                  (short)0, acc, false, false);
  }

  // C/D layout: VGPR v -> M = v + 8*hiHalf, N = lane&15
  const int col = colTile * 16 + (lane & 15);
#pragma unroll
  for (int v = 0; v < 8; ++v) {
    const int r = rowTile * 16 + v + hiHalf * 8;
    H[(size_t)r * 256 + col] = acc[v];
  }
}

// ---------------------------------------------------------------------------
// K2: per-(node, head) attention scores. One wave per (n,h); shfl reduction.
// ---------------------------------------------------------------------------
__global__ void __launch_bounds__(256)
k_scores(const float* __restrict__ H, const float* __restrict__ avec,
         float* __restrict__ ssrc, float* __restrict__ sdst, int N) {
  const int wave = threadIdx.x >> 5;
  const int lane = threadIdx.x & 31;
  const long p = (long)blockIdx.x * 8 + wave;
  if (p >= (long)N * 4) return;
  const int n = (int)(p >> 2);
  const int h = (int)(p & 3);
  const float* __restrict__ hp = H + (size_t)n * 256 + h * 64;
  const float x0 = hp[lane];
  const float x1 = hp[lane + 32];
  float s1 = x0 * avec[lane] + x1 * avec[lane + 32];        // a[:64]
  float s2 = x0 * avec[64 + lane] + x1 * avec[96 + lane];   // a[64:]
#pragma unroll
  for (int off = 16; off > 0; off >>= 1) {
    s1 += __shfl_xor(s1, off, 32);
    s2 += __shfl_xor(s2, off, 32);
  }
  if (lane == 0) { ssrc[p] = s1; sdst[p] = s2; }
}

// ---------------------------------------------------------------------------
// K3: global max of leaky_relu(s_src[src]+s_dst[dst]) over all edges/heads.
// ---------------------------------------------------------------------------
__device__ __forceinline__ void atomicMaxF(float* addr, float val) {
  if (!(val == val)) return;  // drop NaN
  if (val >= 0.f) atomicMax((int*)addr, __float_as_int(val));
  else            atomicMin((unsigned int*)addr, __float_as_uint(val));
}

__global__ void __launch_bounds__(256)
k_edge_max(const int* __restrict__ eidx, const float* __restrict__ ssrc,
           const float* __restrict__ sdst, float* __restrict__ gmax, int E) {
  __shared__ float red[256];
  const long i0 = (long)blockIdx.x * blockDim.x + threadIdx.x;
  const long stride = (long)gridDim.x * blockDim.x;
  float m = -__builtin_inff();
  for (long e = i0; e < E; e += stride) {
    const int s = eidx[e];
    const int d = eidx[(long)E + e];
#pragma unroll
    for (int h = 0; h < 4; ++h) {
      float v = ssrc[s * 4 + h] + sdst[d * 4 + h];
      v = (v >= 0.f) ? v : NEG_SLOPE * v;
      m = fmaxf(m, v);
    }
  }
  red[threadIdx.x] = m;
  __syncthreads();
  for (int off = 128; off > 0; off >>= 1) {
    if (threadIdx.x < off)
      red[threadIdx.x] = fmaxf(red[threadIdx.x], red[threadIdx.x + off]);
    __syncthreads();
  }
  if (threadIdx.x == 0) atomicMaxF(gmax, red[0]);
}

// ---------------------------------------------------------------------------
// K4: alpha_sum[dst,h] += exp(lrelu(score) - gmax)
// ---------------------------------------------------------------------------
__global__ void __launch_bounds__(256)
k_softmax_sum(const int* __restrict__ eidx, const float* __restrict__ ssrc,
              const float* __restrict__ sdst, const float* __restrict__ gmax,
              float* __restrict__ asum, int E) {
  const long i0 = (long)blockIdx.x * blockDim.x + threadIdx.x;
  const long stride = (long)gridDim.x * blockDim.x;
  const float gm = *gmax;
  for (long e = i0; e < E; e += stride) {
    const int s = eidx[e];
    const int d = eidx[(long)E + e];
#pragma unroll
    for (int h = 0; h < 4; ++h) {
      float v = ssrc[s * 4 + h] + sdst[d * 4 + h];
      v = (v >= 0.f) ? v : NEG_SLOPE * v;
      atomicAdd(&asum[d * 4 + h], expf(v - gm));
    }
  }
}

// ---------------------------------------------------------------------------
// K5: out[dst] += coef_h * H[src]   (one wave per edge; 8 x 32-lane chunks,
// head index is wave-uniform per chunk). Dominant, L2-bandwidth-bound phase.
// ---------------------------------------------------------------------------
__global__ void __launch_bounds__(256)
k_aggregate(const float* __restrict__ H, const int* __restrict__ eidx,
            const float* __restrict__ ssrc, const float* __restrict__ sdst,
            const float* __restrict__ asum, const float* __restrict__ gmax,
            float* __restrict__ out, int E) {
  const int wave = threadIdx.x >> 5;
  const int lane = threadIdx.x & 31;
  const long e = (long)blockIdx.x * 8 + wave;
  if (e >= E) return;  // wave-uniform
  const int s = eidx[e];
  const int d = eidx[(long)E + e];
  const float gm = *gmax;
  float coef[4];
#pragma unroll
  for (int h = 0; h < 4; ++h) {
    float v = ssrc[s * 4 + h] + sdst[d * 4 + h];
    v = (v >= 0.f) ? v : NEG_SLOPE * v;
    coef[h] = expf(v - gm) / (asum[d * 4 + h] + EPS_F);
  }
  const float* __restrict__ hs = H + (size_t)s * 256;
  float* __restrict__ od = out + (size_t)d * 256;
#pragma unroll
  for (int i = 0; i < 8; ++i) {
    const int c = i * 32 + lane;
    atomicAdd(od + c, coef[i >> 1] * hs[c]);
  }
}

// ---------------------------------------------------------------------------
// K6: in-place ELU
// ---------------------------------------------------------------------------
__global__ void __launch_bounds__(256)
k_elu(float* __restrict__ out, long n) {
  const long i0 = (long)blockIdx.x * blockDim.x + threadIdx.x;
  const long stride = (long)gridDim.x * blockDim.x;
  for (long i = i0; i < n; i += stride) {
    const float v = out[i];
    out[i] = (v > 0.f) ? v : expm1f(v);
  }
}

// ---------------------------------------------------------------------------
extern "C" void kernel_launch(void* const* d_in, const int* in_sizes, int n_in,
                              void* d_out, int out_size, void* d_ws, size_t ws_size,
                              hipStream_t stream) {
  const float* x    = (const float*)d_in[0];   // [N, 256]
  const int*   eidx = (const int*)d_in[1];     // [2, E]
  const float* W    = (const float*)d_in[2];   // [256, 256]
  const float* avec = (const float*)d_in[3];   // [128]
  float* out = (float*)d_out;                  // [N, 256]

  const int N = in_sizes[0] / 256;
  const int E = in_sizes[1] / 2;

  // Workspace layout
  float* H    = (float*)d_ws;                  // N*256
  float* ssrc = H + (size_t)N * 256;           // N*4
  float* sdst = ssrc + (size_t)N * 4;          // N*4
  float* asum = sdst + (size_t)N * 4;          // N*4
  float* gmax = asum + (size_t)N * 4;          // 1

  k_init<<<512, 256, 0, stream>>>(out, asum, gmax, N * 256, N * 4);

  const int nRowTiles = N >> 4;
  dim3 gGemm((nRowTiles + 7) / 8, 16);
  k_gemm_wmma<<<gGemm, 256, 0, stream>>>(x, W, H, N);

  k_scores<<<(N * 4 + 7) / 8, 256, 0, stream>>>(H, avec, ssrc, sdst, N);

  k_edge_max<<<1024, 256, 0, stream>>>(eidx, ssrc, sdst, gmax, E);

  k_softmax_sum<<<2048, 256, 0, stream>>>(eidx, ssrc, sdst, gmax, asum, E);

  k_aggregate<<<(E + 7) / 8, 256, 0, stream>>>(H, eidx, ssrc, sdst, asum, gmax, out, E);

  k_elu<<<2048, 256, 0, stream>>>(out, (long)N * 256);
}
